// Olmo3MoeAttention_23141283791734
// MI455X (gfx1250) — compile-verified, matrix operand order checked by reading, not run on hardware
//
#include <hip/hip_runtime.h>
#include <hip/hip_bf16.h>

// ---------------------------------------------------------------------------
// Olmo3 attention block for gfx1250 (MI455X): bf16 WMMA everywhere,
// async global->LDS DMA staging (ASYNCcnt) in both GEMM and attention.
// ---------------------------------------------------------------------------

typedef __attribute__((ext_vector_type(16))) __bf16 bf16x16;
typedef __attribute__((ext_vector_type(8)))  float  f32x8;

#define WMMA_BF16(A, Bm, Cm) \
  __builtin_amdgcn_wmma_f32_16x16x32_bf16(false, (A), false, (Bm), (short)0, (Cm), false, false)

#define B_    2
#define S_    2048
#define HID_  2048
#define NH_   32
#define NKV_  8
#define HD_   64
#define SCALE_ 0.125f          // HD^-0.5
#define NEG_BIG (-3.0e38f)

__device__ __forceinline__ unsigned short f2bf(float f) {
  unsigned u = __float_as_uint(f);
  u += 0x7FFFu + ((u >> 16) & 1u);   // round-to-nearest-even
  return (unsigned short)(u >> 16);
}

// Async global->LDS copy of 16 bytes per lane (ASYNCcnt-tracked DMA).
__device__ __forceinline__ void async_ld_lds_b128(const void* gptr, void* lptr) {
  unsigned ldsoff = (unsigned)(unsigned long long)lptr;   // low 32b of generic = LDS offset
  asm volatile("global_load_async_to_lds_b128 %0, %1, off"
               :: "v"(ldsoff), "v"(gptr) : "memory");
}
__device__ __forceinline__ void wait_asynccnt0() {
  asm volatile("s_wait_asynccnt 0x0" ::: "memory");
}

// ---------------------------------------------------------------------------
// f32 -> bf16 bulk convert (4 elems / thread)
// ---------------------------------------------------------------------------
__global__ __launch_bounds__(256) void cvt_f32_bf16(const float* __restrict__ in,
                                                    unsigned short* __restrict__ out,
                                                    int n) {
  int i = (blockIdx.x * 256 + threadIdx.x) * 4;
  if (i + 3 < n) {
    float4 v = *(const float4*)(in + i);
    unsigned r0 = (unsigned)f2bf(v.x) | ((unsigned)f2bf(v.y) << 16);
    unsigned r1 = (unsigned)f2bf(v.z) | ((unsigned)f2bf(v.w) << 16);
    uint2 r; r.x = r0; r.y = r1;
    *(uint2*)(out + i) = r;
  }
}

// ---------------------------------------------------------------------------
// GEMM: C[M,N] (f32) = A[M,K](bf16, row-major) * W[N,K](bf16, row-major)^T
// Block: 256 threads = 8 waves; block tile 128x128; wave tile 32x64 (4x2 wave
// grid); K-step 32 -> 8 WMMAs per wave per step. Tiles staged with async
// global->LDS DMA (no VGPR round-trip).
// ---------------------------------------------------------------------------
__global__ __launch_bounds__(256) void gemm_bf16(const unsigned short* __restrict__ A,
                                                 const unsigned short* __restrict__ W,
                                                 float* __restrict__ C,
                                                 int M, int N, int K) {
  __shared__ alignas(32) unsigned short As[128 * 48];   // 128 rows x (32+16 pad)
  __shared__ alignas(32) unsigned short Bs[128 * 48];

  const int tid  = threadIdx.x;
  const int w    = tid >> 5;
  const int lane = tid & 31;
  const int lo   = lane & 15;
  const int hi   = lane >> 4;
  const int mw   = (w >> 1) * 32;       // 0,32,64,96
  const int nw   = (w & 1) * 64;        // 0,64
  const int gm0  = blockIdx.y * 128;
  const int gn0  = blockIdx.x * 128;

  f32x8 acc[2][4];
#pragma unroll
  for (int i = 0; i < 2; ++i)
#pragma unroll
    for (int j = 0; j < 4; ++j) acc[i][j] = (f32x8){};

  const int ldRow = tid >> 1;           // 0..127
  const int ldCol = (tid & 1) * 16;     // 0,16

  for (int k0 = 0; k0 < K; k0 += 32) {
    __syncthreads();
    const unsigned short* aSrc = &A[(size_t)(gm0 + ldRow) * K + k0 + ldCol];
    const unsigned short* bSrc = &W[(size_t)(gn0 + ldRow) * K + k0 + ldCol];
    async_ld_lds_b128(aSrc,     &As[ldRow * 48 + ldCol]);
    async_ld_lds_b128(aSrc + 8, &As[ldRow * 48 + ldCol + 8]);
    async_ld_lds_b128(bSrc,     &Bs[ldRow * 48 + ldCol]);
    async_ld_lds_b128(bSrc + 8, &Bs[ldRow * 48 + ldCol + 8]);
    // prefetch next K-slice (speculative; OOB prefetches are dropped)
    __builtin_prefetch(aSrc + 32, 0, 3);
    __builtin_prefetch(bSrc + 32, 0, 3);
    wait_asynccnt0();
    __syncthreads();

    bf16x16 a0 = *(const bf16x16*)&As[(mw + lo) * 48 + hi * 16];
    bf16x16 a1 = *(const bf16x16*)&As[(mw + 16 + lo) * 48 + hi * 16];
#pragma unroll
    for (int j = 0; j < 4; ++j) {
      bf16x16 bj = *(const bf16x16*)&Bs[(nw + j * 16 + lo) * 48 + hi * 16];
      acc[0][j] = WMMA_BF16(a0, bj, acc[0][j]);
      acc[1][j] = WMMA_BF16(a1, bj, acc[1][j]);
    }
  }

#pragma unroll
  for (int i = 0; i < 2; ++i)
#pragma unroll
    for (int j = 0; j < 4; ++j)
#pragma unroll
      for (int p = 0; p < 8; ++p) {
        int r = gm0 + mw + i * 16 + p + 8 * hi;
        int c = gn0 + nw + j * 16 + lo;
        C[(size_t)r * N + c] = acc[i][j][p];
      }
}

// ---------------------------------------------------------------------------
// RMSNorm + RoPE over head rows of 64. One wave per (b,s,h) row; lane owns
// elements d=lane and d=lane+32 (matches rotate_half split).
// In : X [B*S, nH*64] f32.  Out: bf16 [B, nH, S, 64].
// ---------------------------------------------------------------------------
__global__ __launch_bounds__(256) void norm_rope(const float* __restrict__ X,
                                                 const float* __restrict__ wvec,
                                                 const float* __restrict__ cb,
                                                 const float* __restrict__ sb,
                                                 unsigned short* __restrict__ Out,
                                                 int nH) {
  const int w    = threadIdx.x >> 5;
  const int lane = threadIdx.x & 31;
  const int r    = blockIdx.x * 8 + w;       // row over B*S*nH
  const int h    = r % nH;
  const int bs   = r / nH;                   // b*S + s

  const float* x = X + ((size_t)bs * nH + h) * 64;
  float e0 = x[lane];
  float e1 = x[lane + 32];

  float ss = e0 * e0 + e1 * e1;
  ss += __shfl_xor(ss, 1);
  ss += __shfl_xor(ss, 2);
  ss += __shfl_xor(ss, 4);
  ss += __shfl_xor(ss, 8);
  ss += __shfl_xor(ss, 16);
  float rinv = rsqrtf(ss * (1.0f / 64.0f) + 1e-6f);

  float n0 = wvec[lane] * e0 * rinv;
  float n1 = wvec[lane + 32] * e1 * rinv;

  const float* cp = cb + (size_t)bs * 64;
  const float* sp = sb + (size_t)bs * 64;
  float o0 = n0 * cp[lane]      - n1 * sp[lane];        // d < 32: x*c - x2*s
  float o1 = n1 * cp[lane + 32] + n0 * sp[lane + 32];   // d >=32: x*c + x1*s

  const int b = bs / S_;
  const int s = bs % S_;
  unsigned short* op = Out + (((size_t)b * nH + h) * S_ + s) * 64;
  op[lane]      = f2bf(o0);
  op[lane + 32] = f2bf(o1);
}

// ---------------------------------------------------------------------------
// V convert + transpose: f32 [B*S, NKV*64] -> bf16 [B, NKV, 64, S]
// ---------------------------------------------------------------------------
__global__ __launch_bounds__(256) void v_convert(const float* __restrict__ V,
                                                 unsigned short* __restrict__ Vt) {
  int i  = blockIdx.x * 256 + threadIdx.x;   // over B*S*NKV*64
  int bs = i >> 9;
  int hd = i & 511;
  int h  = hd >> 6;
  int d  = hd & 63;
  int b  = bs / S_;
  int s  = bs % S_;
  Vt[(((size_t)b * NKV_ + h) * 64 + d) * S_ + s] = f2bf(V[i]);
}

// ---------------------------------------------------------------------------
// Causal flash attention (GQA). Block = 4 waves x 16 q-rows = 64 q-rows.
// Q bf16 [B,NH,S,64], K bf16 [B,NKV,S,64], Vt bf16 [B,NKV,64,S].
// K/V tiles staged to LDS via ASYNC global->LDS DMA (ASYNCcnt).
// Output bf16 [B*S, NH*64] (ready as GEMM A for Wo projection).
// ---------------------------------------------------------------------------
__global__ __launch_bounds__(128) void attn_kernel(const unsigned short* __restrict__ Q,
                                                   const unsigned short* __restrict__ Kc,
                                                   const unsigned short* __restrict__ Vt,
                                                   unsigned short* __restrict__ O) {
  __shared__ alignas(32) unsigned short Ks[32 * 80];      // [key][d], 64+16 pad
  __shared__ alignas(32) unsigned short VTs[64 * 48];     // [d][key], 32+16 pad
  __shared__ alignas(32) unsigned short Ps[4 * 16 * 32];  // per-wave P staging

  const int tid  = threadIdx.x;
  const int w    = tid >> 5;
  const int lane = tid & 31;
  const int lo   = lane & 15;
  const int hi   = lane >> 4;

  const int bh    = blockIdx.x;
  const int b     = bh >> 5;          // / NH
  const int h     = bh & 31;
  const int hk    = h >> 2;           // GQA: 4 q-heads per kv-head
  const int qBase = blockIdx.y * 64;
  const int qRow0 = qBase + w * 16;

  // Q fragments (persistent in registers): 16 rows x 64 d -> 2 A-frags (K=32)
  const unsigned short* qptr =
      Q + ((size_t)(b * NH_ + h) * S_ + (qRow0 + lo)) * HD_;
  bf16x16 aQ0 = *(const bf16x16*)(qptr + hi * 16);
  bf16x16 aQ1 = *(const bf16x16*)(qptr + 32 + hi * 16);

  f32x8 o0 = {}, o1 = {}, o2 = {}, o3 = {};
  float m[8], l[8];
#pragma unroll
  for (int p = 0; p < 8; ++p) { m[p] = -1e30f; l[p] = 0.0f; }

  const unsigned short* kbase = Kc + (size_t)(b * NKV_ + hk) * S_ * HD_;
  const unsigned short* vbase = Vt + (size_t)(b * NKV_ + hk) * HD_ * S_;

  // per-thread staging coordinates
  const int kKey = tid >> 2, kD = (tid & 3) * 16;   // K tile: [32 keys][64 d]
  const int vD = tid >> 1, vHalf = tid & 1;         // V tile: [64 d][32 keys]

  const int nChunks = (qBase >> 5) + 2;     // keys [0, qBase+64)
  for (int kc = 0; kc < nChunks; ++kc) {
    const int k0 = kc * 32;
    __syncthreads();   // previous iteration's LDS reads done before overwrite
    // async DMA: K chunk [32 keys][64 d] and V chunk transposed [64 d][32 keys]
    {
      const unsigned short* src = kbase + (size_t)(k0 + kKey) * HD_ + kD;
      async_ld_lds_b128(src,     &Ks[kKey * 80 + kD]);
      async_ld_lds_b128(src + 8, &Ks[kKey * 80 + kD + 8]);
    }
    {
      const unsigned short* src = vbase + (size_t)vD * S_ + k0 + vHalf * 16;
      async_ld_lds_b128(src,     &VTs[vD * 48 + vHalf * 16]);
      async_ld_lds_b128(src + 8, &VTs[vD * 48 + vHalf * 16 + 8]);
    }
    wait_asynccnt0();
    __syncthreads();

    // scores: two 16x16 tiles (keys k0..k0+15 and k0+16..k0+31), contraction d=64
    f32x8 s0 = {}, s1 = {};
    {
      bf16x16 bk;
      bk = *(const bf16x16*)&Ks[(lo)      * 80 + hi * 16];
      s0 = WMMA_BF16(aQ0, bk, s0);
      bk = *(const bf16x16*)&Ks[(lo)      * 80 + 32 + hi * 16];
      s0 = WMMA_BF16(aQ1, bk, s0);
      bk = *(const bf16x16*)&Ks[(16 + lo) * 80 + hi * 16];
      s1 = WMMA_BF16(aQ0, bk, s1);
      bk = *(const bf16x16*)&Ks[(16 + lo) * 80 + 32 + hi * 16];
      s1 = WMMA_BF16(aQ1, bk, s1);
    }

    const int key0 = k0 + lo;
    const int key1 = k0 + 16 + lo;
#pragma unroll
    for (int p = 0; p < 8; ++p) {
      const int qrow = qRow0 + p + 8 * hi;
      float v0 = s0[p] * SCALE_;
      float v1 = s1[p] * SCALE_;
      if (key0 > qrow) v0 = NEG_BIG;
      if (key1 > qrow) v1 = NEG_BIG;

      float mx = fmaxf(v0, v1);
      mx = fmaxf(mx, __shfl_xor(mx, 1));
      mx = fmaxf(mx, __shfl_xor(mx, 2));
      mx = fmaxf(mx, __shfl_xor(mx, 4));
      mx = fmaxf(mx, __shfl_xor(mx, 8));

      float mn    = fmaxf(m[p], mx);
      float alpha = __expf(m[p] - mn);
      m[p] = mn;
      float p0 = __expf(v0 - mn);
      float p1 = __expf(v1 - mn);
      float rs = p0 + p1;
      rs += __shfl_xor(rs, 1);
      rs += __shfl_xor(rs, 2);
      rs += __shfl_xor(rs, 4);
      rs += __shfl_xor(rs, 8);
      l[p] = l[p] * alpha + rs;
      o0[p] *= alpha; o1[p] *= alpha; o2[p] *= alpha; o3[p] *= alpha;

      const int prow = p + 8 * hi;
      Ps[(w * 16 + prow) * 32 + lo]      = f2bf(p0);
      Ps[(w * 16 + prow) * 32 + 16 + lo] = f2bf(p1);
    }

    // P * V : contraction over 32 keys, 4 output d-tiles of 16
    bf16x16 aP = *(const bf16x16*)&Ps[(w * 16 + lo) * 32 + hi * 16];
    {
      bf16x16 bv;
      bv = *(const bf16x16*)&VTs[(lo)      * 48 + hi * 16];
      o0 = WMMA_BF16(aP, bv, o0);
      bv = *(const bf16x16*)&VTs[(16 + lo) * 48 + hi * 16];
      o1 = WMMA_BF16(aP, bv, o1);
      bv = *(const bf16x16*)&VTs[(32 + lo) * 48 + hi * 16];
      o2 = WMMA_BF16(aP, bv, o2);
      bv = *(const bf16x16*)&VTs[(48 + lo) * 48 + hi * 16];
      o3 = WMMA_BF16(aP, bv, o3);
    }
  }

#pragma unroll
  for (int p = 0; p < 8; ++p) {
    const int qrow = qRow0 + p + 8 * hi;
    float invl = 1.0f / l[p];
    size_t rowoff = ((size_t)b * S_ + qrow) * (size_t)(NH_ * HD_) + (size_t)h * HD_;
    O[rowoff + lo]      = f2bf(o0[p] * invl);
    O[rowoff + 16 + lo] = f2bf(o1[p] * invl);
    O[rowoff + 32 + lo] = f2bf(o2[p] * invl);
    O[rowoff + 48 + lo] = f2bf(o3[p] * invl);
  }
}

// ---------------------------------------------------------------------------
// Host-side orchestration
// ---------------------------------------------------------------------------
extern "C" void kernel_launch(void* const* d_in, const int* in_sizes, int n_in,
                              void* d_out, int out_size, void* d_ws, size_t ws_size,
                              hipStream_t stream) {
  (void)in_sizes; (void)n_in; (void)out_size; (void)ws_size;

  const float* hidden = (const float*)d_in[0];
  const float* cosb   = (const float*)d_in[1];
  const float* sinb   = (const float*)d_in[2];
  const float* Wq     = (const float*)d_in[3];
  const float* Wk     = (const float*)d_in[4];
  const float* Wv     = (const float*)d_in[5];
  const float* Wo     = (const float*)d_in[6];
  const float* qnw    = (const float*)d_in[7];
  const float* knw    = (const float*)d_in[8];
  float* out = (float*)d_out;

  const int M = B_ * S_;          // 4096
  char* base = (char*)d_ws;
  size_t off = 0;
  auto alloc = [&](size_t bytes) -> void* {
    void* p = base + off;
    off = (off + bytes + 255) & ~(size_t)255;
    return p;
  };

  unsigned short* Xbf  = (unsigned short*)alloc((size_t)M * HID_ * 2);
  unsigned short* Wqbf = (unsigned short*)alloc((size_t)HID_ * HID_ * 2);
  unsigned short* Wkbf = (unsigned short*)alloc((size_t)NKV_ * HD_ * HID_ * 2);
  unsigned short* Wvbf = (unsigned short*)alloc((size_t)NKV_ * HD_ * HID_ * 2);
  unsigned short* Wobf = (unsigned short*)alloc((size_t)HID_ * HID_ * 2);
  float* qf = (float*)alloc((size_t)M * HID_ * 4);
  float* kf = (float*)alloc((size_t)M * NKV_ * HD_ * 4);
  float* vf = (float*)alloc((size_t)M * NKV_ * HD_ * 4);
  unsigned short* Qbf = (unsigned short*)alloc((size_t)M * HID_ * 2);
  unsigned short* Kbf = (unsigned short*)alloc((size_t)M * NKV_ * HD_ * 2);
  unsigned short* Vtb = (unsigned short*)alloc((size_t)M * NKV_ * HD_ * 2);
  unsigned short* Abf = (unsigned short*)alloc((size_t)M * HID_ * 2);

  // 1) precision conversion of activations + weights
  cvt_f32_bf16<<<(M * HID_) / 1024, 256, 0, stream>>>(hidden, Xbf, M * HID_);
  cvt_f32_bf16<<<(HID_ * HID_) / 1024, 256, 0, stream>>>(Wq, Wqbf, HID_ * HID_);
  cvt_f32_bf16<<<(NKV_ * HD_ * HID_) / 1024, 256, 0, stream>>>(Wk, Wkbf, NKV_ * HD_ * HID_);
  cvt_f32_bf16<<<(NKV_ * HD_ * HID_) / 1024, 256, 0, stream>>>(Wv, Wvbf, NKV_ * HD_ * HID_);
  cvt_f32_bf16<<<(HID_ * HID_) / 1024, 256, 0, stream>>>(Wo, Wobf, HID_ * HID_);

  // 2) QKV projections (bf16 WMMA, f32 out)
  gemm_bf16<<<dim3(HID_ / 128, M / 128), 256, 0, stream>>>(Xbf, Wqbf, qf, M, HID_, HID_);
  gemm_bf16<<<dim3((NKV_ * HD_) / 128, M / 128), 256, 0, stream>>>(Xbf, Wkbf, kf, M, NKV_ * HD_, HID_);
  gemm_bf16<<<dim3((NKV_ * HD_) / 128, M / 128), 256, 0, stream>>>(Xbf, Wvbf, vf, M, NKV_ * HD_, HID_);

  // 3) QK RMSNorm + RoPE (-> head-major bf16); V transpose-convert
  norm_rope<<<(M * NH_) / 8, 256, 0, stream>>>(qf, qnw, cosb, sinb, Qbf, NH_);
  norm_rope<<<(M * NKV_) / 8, 256, 0, stream>>>(kf, knw, cosb, sinb, Kbf, NKV_);
  v_convert<<<(M * NKV_ * HD_) / 256, 256, 0, stream>>>(vf, Vtb);

  // 4) causal flash attention (async LDS staging)
  attn_kernel<<<dim3(B_ * NH_, S_ / 64), 128, 0, stream>>>(Qbf, Kbf, Vtb, Abf);

  // 5) output projection
  gemm_bf16<<<dim3(HID_ / 128, M / 128), 256, 0, stream>>>(Abf, Wobf, out, M, HID_, HID_);
}
